// DeepGCN_Base_27891517620234
// MI455X (gfx1250) — matrix operand order repaired and checked by reading, not compiled, not run on hardware
//
#include <hip/hip_runtime.h>

#define N_GRAPHS 16
#define NODES    1024
#define DEG      16
#define IN_CH    16
#define CHN      256
#define NB       28
#define FUS_IN   7168
#define POOL_DIM 1024
#define KSEL     717
#define NTOT     16384
#define EPSB     1e-5f

typedef __attribute__((ext_vector_type(16))) __bf16       v16bf;
typedef __attribute__((ext_vector_type(8)))  float        v8f;
typedef __attribute__((ext_vector_type(8)))  unsigned int v8u;

__device__ __forceinline__ unsigned short f2bf(float f) {
  union { float f; unsigned int u; } v; v.f = f;
  unsigned int r = v.u + 0x7FFFu + ((v.u >> 16) & 1u);   // round-to-nearest-even
  return (unsigned short)(r >> 16);
}

// batched transpose + convert: o[b][n][k] = bf16(w[b][k][n])
__global__ void cvt_bf16_t_kernel(const float* __restrict__ w,
                                  unsigned short* __restrict__ o,
                                  int Kd, int Nd) {
  long mat = blockIdx.y;
  long i = (long)blockIdx.x * blockDim.x + threadIdx.x;
  if (i >= (long)Kd * Nd) return;
  int n = (int)(i / Kd), k = (int)(i % Kd);
  o[mat * (long)Kd * Nd + i] = f2bf(w[mat * (long)Kd * Nd + (long)k * Nd + n]);
}

// zin[n] = [ h[n] | max_j h[src[16n+j]] - h[n] ]  (bf16); edges per node contiguous,
// so segment_max(h[src]-h[dst]) == (max_j h[src_j]) - h[n] per channel.
__global__ void build_zin_kernel(const float* __restrict__ h, int hStride, int C,
                                 const int* __restrict__ src,
                                 unsigned short* __restrict__ zin) {
  int n = blockIdx.x;
  int c = threadIdx.x;
  __shared__ int s[DEG];
  if (c < DEG) s[c] = src[n * DEG + c];
  __syncthreads();
  if (c >= C) return;
  float hv = h[(long)n * hStride + c];
  float m = -3.4e38f;
#pragma unroll
  for (int j = 0; j < DEG; ++j)
    m = fmaxf(m, h[(long)s[j] * hStride + c]);
  zin[(long)n * (2 * C) + c]     = f2bf(hv);
  zin[(long)n * (2 * C) + C + c] = f2bf(m - hv);
}

// C[M x N] = relu(A[M x K](bf16,row-major) @ BT[N x K](bf16,row-major)^T + bias)
// 256 threads = 8 waves; block tile 128x128; wave tile 32x64: 8 WMMAs per k-step
// from 6 fragments (2 A x 4 B) -> 1.5 ds_load_b128 per WMMA. Double-buffered LDS,
// one barrier per k-step, pointer-increment addressing, prefetch 2 tiles ahead.
__global__ __launch_bounds__(256)
void gemm_bf16_relu_kernel(const unsigned short* __restrict__ A, int lda,
                           const unsigned short* __restrict__ BT, int ldb,
                           const float* __restrict__ bias,
                           float* __restrict__ C, int ldc, int K) {
  __shared__ uint4 As4[2][512];   // 128 rows x 32 bf16 per buffer
  __shared__ uint4 Bs4[2][512];   // 128 cols x 32 bf16 per buffer (BT rows)

  const int tid  = threadIdx.x;
  const int wave = tid >> 5;
  const int lane = tid & 31;
  const int half = lane >> 4;
  const int lr   = lane & 15;
  const int wr   = wave >> 1;          // 0..3  (32-row band)
  const int wc   = wave & 1;           // 0..1  (64-col band)
  const long m0  = (long)blockIdx.x * 128;
  const long n0  = (long)blockIdx.y * 128;

  // staging: thread owns 32B of one row: row = tid>>1, half-row = tid&1
  const int ldRow = tid >> 1;
  const int ldH   = tid & 1;
  const unsigned short* aPtr = A  + (m0 + ldRow) * (long)lda + ldH * 16;
  const unsigned short* bPtr = BT + (n0 + ldRow) * (long)ldb + ldH * 16;
  const int sIdx = ldRow * 4 + ldH * 2;

  v8f acc[2][4] = {};

  {
    uint4 a0 = *(const uint4*)aPtr, a1 = *(const uint4*)(aPtr + 8);
    uint4 b0 = *(const uint4*)bPtr, b1 = *(const uint4*)(bPtr + 8);
    As4[0][sIdx] = a0; As4[0][sIdx + 1] = a1;
    Bs4[0][sIdx] = b0; Bs4[0][sIdx + 1] = b1;
  }
  __syncthreads();

  int buf = 0;
  for (int k0 = 0; k0 < K; k0 += 32) {
    if (k0 + 32 < K) {
      uint4 a0 = *(const uint4*)(aPtr + k0 + 32), a1 = *(const uint4*)(aPtr + k0 + 40);
      uint4 b0 = *(const uint4*)(bPtr + k0 + 32), b1 = *(const uint4*)(bPtr + k0 + 40);
      As4[buf ^ 1][sIdx] = a0; As4[buf ^ 1][sIdx + 1] = a1;
      Bs4[buf ^ 1][sIdx] = b0; Bs4[buf ^ 1][sIdx + 1] = b1;
      if (k0 + 64 < K) {
        __builtin_prefetch(aPtr + k0 + 64, 0, 1);   // global_prefetch_b8
        __builtin_prefetch(bPtr + k0 + 64, 0, 1);
      }
    }
    const unsigned int* As32 = (const unsigned int*)As4[buf];
    const unsigned int* Bs32 = (const unsigned int*)Bs4[buf];

    // A frag (16-bit 16x32): lane<16 row M=lr K{0..7,16..23}; lane>=16 K{8..15,24..31}
    v8u au[2];
#pragma unroll
    for (int t = 0; t < 2; ++t) {
      const unsigned int* p = &As32[(wr * 32 + t * 16 + lr) * 16 + half * 4];
#pragma unroll
      for (int v = 0; v < 4; ++v) { au[t][v] = p[v]; au[t][4 + v] = p[8 + v]; }
    }
    // B frag (32x16): lane<16 col N=lr K{0..7,16..23}; lane>=16 K{8..15,24..31} (K-major in LDS)
    v8u bu[4];
#pragma unroll
    for (int u = 0; u < 4; ++u) {
      const unsigned int* q = &Bs32[(wc * 64 + u * 16 + lr) * 16 + half * 4];
#pragma unroll
      for (int v = 0; v < 4; ++v) { bu[u][v] = q[v]; bu[u][4 + v] = q[8 + v]; }
    }

#pragma unroll
    for (int t = 0; t < 2; ++t) {
      v16bf a = __builtin_bit_cast(v16bf, au[t]);
#pragma unroll
      for (int u = 0; u < 4; ++u) {
        v16bf b = __builtin_bit_cast(v16bf, bu[u]);
        acc[t][u] = __builtin_amdgcn_wmma_f32_16x16x32_bf16(
            false, a, false, b, (short)0, acc[t][u], false, false);
      }
    }
    __syncthreads();
    buf ^= 1;
  }

  // C layout: VGPR v -> row +8*half+v, col = lane%16
#pragma unroll
  for (int t = 0; t < 2; ++t) {
#pragma unroll
    for (int u = 0; u < 4; ++u) {
      long col = n0 + wc * 64 + u * 16 + lr;
      float bv = bias[col];
#pragma unroll
      for (int v = 0; v < 8; ++v) {
        long row = m0 + wr * 32 + t * 16 + half * 8 + v;
        float x = acc[t][u][v] + bv;
        C[row * ldc + col] = x > 0.f ? x : 0.f;
      }
    }
  }
}

// deterministic two-stage BN stats over NTOT rows
__global__ void bn_partial_kernel(const float* __restrict__ t, int C,
                                  float* __restrict__ ps, float* __restrict__ pq) {
  int rb = blockIdx.x;               // 256 row blocks x 64 rows
  for (int cb = 0; cb < C; cb += 256) {
    int c = cb + threadIdx.x;
    float s = 0.f, q = 0.f;
    for (int r = 0; r < 64; ++r) {
      float v = t[(long)(rb * 64 + r) * C + c];
      s += v; q += v * v;
    }
    ps[(long)rb * C + c] = s;
    pq[(long)rb * C + c] = q;
  }
}

__global__ void bn_finalize_kernel(const float* __restrict__ ps,
                                   const float* __restrict__ pq, int C,
                                   float* __restrict__ mu, float* __restrict__ var) {
  int c = blockIdx.x * 256 + threadIdx.x;
  if (c >= C) return;
  float s = 0.f, q = 0.f;
  for (int p = 0; p < 256; ++p) { s += ps[(long)p * C + c]; q += pq[(long)p * C + c]; }
  float m = s / (float)NTOT;
  mu[c]  = m;
  var[c] = q / (float)NTOT - m * m;
}

// y = g*(t-mu)*rsqrt(var+eps)+be (+residual); writes f32 slice and optional bf16 mirror
__global__ void bn_apply_kernel(const float* __restrict__ t, int C,
                                const float* __restrict__ mu, const float* __restrict__ var,
                                const float* __restrict__ g, const float* __restrict__ be,
                                const float* __restrict__ res, int resStride,
                                float* __restrict__ outF, int outStride, int outOfs,
                                unsigned short* __restrict__ outB) {
  long i = (long)blockIdx.x * blockDim.x + threadIdx.x;
  long n = i / C; int c = (int)(i % C);
  float v = g[c] * (t[i] - mu[c]) * rsqrtf(var[c] + EPSB) + be[c];
  if (res) v += res[n * resStride + c];
  long oi = n * outStride + outOfs + c;
  outF[oi] = v;
  if (outB) outB[oi] = f2bf(v);
}

__global__ void wnorm_kernel(const float* __restrict__ w, float* __restrict__ invn) {
  __shared__ float sh[256];
  float s = 0.f;
  for (int i = threadIdx.x; i < POOL_DIM; i += 256) s += w[i] * w[i];
  sh[threadIdx.x] = s; __syncthreads();
  for (int o = 128; o > 0; o >>= 1) {
    if (threadIdx.x < o) sh[threadIdx.x] += sh[threadIdx.x + o];
    __syncthreads();
  }
  if (threadIdx.x == 0) invn[0] = rsqrtf(sh[0]);
}

__global__ void score_kernel(const float* __restrict__ fus, const float* __restrict__ w,
                             const float* __restrict__ invn, float* __restrict__ score) {
  int n = blockIdx.x;
  __shared__ float sh[256];
  float s = 0.f;
  for (int i = threadIdx.x; i < POOL_DIM; i += 256)
    s += fus[(long)n * POOL_DIM + i] * w[i];
  sh[threadIdx.x] = s; __syncthreads();
  for (int o = 128; o > 0; o >>= 1) {
    if (threadIdx.x < o) sh[threadIdx.x] += sh[threadIdx.x + o];
    __syncthreads();
  }
  if (threadIdx.x == 0) score[n] = tanhf(sh[0] * invn[0]);
}

// per-graph bitonic sort of 1024 scores (descending, tie-break smaller index)
__global__ void topk_sort_kernel(const float* __restrict__ score,
                                 float* __restrict__ vals, int* __restrict__ nidx) {
  int g = blockIdx.x;
  __shared__ float sc[NODES];
  __shared__ int   si[NODES];
  for (int i = threadIdx.x; i < NODES; i += blockDim.x) {
    sc[i] = score[g * NODES + i];
    si[i] = i;
  }
  for (int k = 2; k <= NODES; k <<= 1) {
    for (int j = k >> 1; j > 0; j >>= 1) {
      __syncthreads();
      for (int i = threadIdx.x; i < NODES; i += blockDim.x) {
        int ixj = i ^ j;
        if (ixj > i) {
          bool desc = ((i & k) == 0);
          float a = sc[i], b = sc[ixj];
          int ia = si[i], ib = si[ixj];
          bool aFirst = (a > b) || (a == b && ia < ib);
          if (desc ? !aFirst : aFirst) {
            sc[i] = b; sc[ixj] = a; si[i] = ib; si[ixj] = ia;
          }
        }
      }
    }
  }
  __syncthreads();
  for (int kk = threadIdx.x; kk < KSEL; kk += blockDim.x) {
    vals[g * KSEL + kk] = sc[kk];
    nidx[g * KSEL + kk] = g * NODES + si[kk];
  }
}

// fused max / mean / min over the K selected (score-scaled) rows
__global__ void pool_stats_kernel(const float* __restrict__ fus,
                                  const float* __restrict__ vals,
                                  const int* __restrict__ nidx,
                                  float* __restrict__ gfeat) {
  int g = blockIdx.x;
  int c = blockIdx.y * 256 + threadIdx.x;
  float mx = -3.4e38f, mn = 3.4e38f, s = 0.f;
  for (int k = 0; k < KSEL; ++k) {
    float v = fus[(long)nidx[g * KSEL + k] * POOL_DIM + c] * vals[g * KSEL + k];
    mx = fmaxf(mx, v); mn = fminf(mn, v); s += v;
  }
  gfeat[(long)g * 3072 + c]        = mx;
  gfeat[(long)g * 3072 + 1024 + c] = s / (float)KSEL;
  gfeat[(long)g * 3072 + 2048 + c] = mn;
}

__global__ void lin_relu_kernel(const float* __restrict__ X, int Kdim,
                                const float* __restrict__ W, const float* __restrict__ b,
                                float* __restrict__ Y, int Nout, int doRelu) {
  int i = blockIdx.x * blockDim.x + threadIdx.x;
  int g = i / Nout, j = i % Nout;
  if (g >= N_GRAPHS) return;
  float s = b[j];
  for (int k = 0; k < Kdim; ++k) s += X[g * Kdim + k] * W[k * Nout + j];
  Y[i] = (doRelu && s < 0.f) ? 0.f : s;
}

__global__ void bn16_kernel(const float* __restrict__ X, int C,
                            const float* __restrict__ gg, const float* __restrict__ be,
                            float* __restrict__ Y) {
  int j = blockIdx.x * blockDim.x + threadIdx.x;
  if (j >= C) return;
  float s = 0.f, q = 0.f;
  for (int g = 0; g < N_GRAPHS; ++g) { float v = X[g * C + j]; s += v; q += v * v; }
  float m = s / 16.f, var = q / 16.f - m * m;
  float inv = rsqrtf(var + EPSB);
  for (int g = 0; g < N_GRAPHS; ++g)
    Y[g * C + j] = gg[j] * (X[g * C + j] - m) * inv + be[j];
}

extern "C" void kernel_launch(void* const* d_in, const int* in_sizes, int n_in,
                              void* d_out, int out_size, void* d_ws, size_t ws_size,
                              hipStream_t stream) {
  (void)in_sizes; (void)n_in; (void)out_size; (void)ws_size;
  const float* x      = (const float*)d_in[0];
  const int*   src    = (const int*)d_in[1];          // edge_index row 0
  const float* W_head = (const float*)d_in[2];
  const float* b_head = (const float*)d_in[3];
  const float* g_head = (const float*)d_in[4];
  const float* be_head= (const float*)d_in[5];
  const float* W_blk  = (const float*)d_in[6];
  const float* b_blk  = (const float*)d_in[7];
  const float* g_blk  = (const float*)d_in[8];
  const float* be_blk = (const float*)d_in[9];
  const float* W_fus  = (const float*)d_in[10];
  const float* b_fus  = (const float*)d_in[11];
  const float* g_fus  = (const float*)d_in[12];
  const float* be_fus = (const float*)d_in[13];
  const float* w_pool = (const float*)d_in[14];
  const float* W_p1   = (const float*)d_in[15];
  const float* b_p1   = (const float*)d_in[16];
  const float* g_p1   = (const float*)d_in[17];
  const float* be_p1  = (const float*)d_in[18];
  const float* W_p2   = (const float*)d_in[19];
  const float* b_p2   = (const float*)d_in[20];
  const float* g_p2   = (const float*)d_in[21];
  const float* be_p2  = (const float*)d_in[22];
  const float* W_p3   = (const float*)d_in[23];
  const float* b_p3   = (const float*)d_in[24];

  char* ws = (char*)d_ws;
  size_t off = 0;
  auto take = [&](size_t bytes) -> char* {
    char* p = ws + off;
    off = (off + bytes + 255) & ~(size_t)255;
    return p;
  };

  float*          fCat  = (float*)take((size_t)NTOT * FUS_IN * 4);
  unsigned short* fCatB = (unsigned short*)take((size_t)NTOT * FUS_IN * 2);
  unsigned short* zin   = (unsigned short*)take((size_t)NTOT * 512 * 2);
  float*          tmp   = (float*)take((size_t)NTOT * 1024 * 4);
  float*          fus   = (float*)take((size_t)NTOT * 1024 * 4);
  unsigned short* WtH   = (unsigned short*)take((size_t)32 * 256 * 2);          // [256 x 32]
  unsigned short* WtB   = (unsigned short*)take((size_t)27 * 512 * 256 * 2);    // 27 x [256 x 512]
  unsigned short* WtF   = (unsigned short*)take((size_t)FUS_IN * 1024 * 2);     // [1024 x 7168]
  float*          ps    = (float*)take((size_t)256 * 1024 * 4);
  float*          pq    = (float*)take((size_t)256 * 1024 * 4);
  float*          mu    = (float*)take(1024 * 4);
  float*          var   = (float*)take(1024 * 4);
  float*          invn  = (float*)take(256);
  float*          score = (float*)take((size_t)NTOT * 4);
  float*          vals  = (float*)take((size_t)N_GRAPHS * KSEL * 4);
  int*            nidx  = (int*)take((size_t)N_GRAPHS * KSEL * 4);
  float*          gfeat = (float*)take((size_t)N_GRAPHS * 3072 * 4);
  float*          h1a   = (float*)take((size_t)N_GRAPHS * 512 * 4);
  float*          h1b   = (float*)take((size_t)N_GRAPHS * 512 * 4);
  float*          h2a   = (float*)take((size_t)N_GRAPHS * 256 * 4);
  float*          h2b   = (float*)take((size_t)N_GRAPHS * 256 * 4);

  // weights -> bf16, transposed to [N x K] so GEMM B-staging is b128 loads/stores
  cvt_bf16_t_kernel<<<dim3((32*256 + 255)/256, 1), 256, 0, stream>>>(W_head, WtH, 32, 256);
  cvt_bf16_t_kernel<<<dim3((512*256 + 255)/256, 27), 256, 0, stream>>>(W_blk, WtB, 512, 256);
  cvt_bf16_t_kernel<<<dim3((FUS_IN*1024 + 255)/256, 1), 256, 0, stream>>>(W_fus, WtF, FUS_IN, 1024);

  // head layer (K = 2*IN_CH = 32)
  build_zin_kernel<<<NTOT, IN_CH, 0, stream>>>(x, IN_CH, IN_CH, src, zin);
  gemm_bf16_relu_kernel<<<dim3(NTOT/128, CHN/128), 256, 0, stream>>>(
      zin, 2*IN_CH, WtH, 2*IN_CH, b_head, tmp, CHN, 2*IN_CH);
  bn_partial_kernel<<<256, 256, 0, stream>>>(tmp, CHN, ps, pq);
  bn_finalize_kernel<<<1, 256, 0, stream>>>(ps, pq, CHN, mu, var);
  bn_apply_kernel<<<(NTOT*CHN)/256, 256, 0, stream>>>(
      tmp, CHN, mu, var, g_head, be_head, nullptr, 0, fCat, FUS_IN, 0, fCatB);

  // 27 residual blocks (K = 2*CHN = 512)
  for (int i = 1; i < NB; ++i) {
    const float* hprev = fCat + (i - 1) * CHN;            // stride FUS_IN
    build_zin_kernel<<<NTOT, CHN, 0, stream>>>(hprev, FUS_IN, CHN, src, zin);
    gemm_bf16_relu_kernel<<<dim3(NTOT/128, CHN/128), 256, 0, stream>>>(
        zin, 2*CHN, WtB + (size_t)(i - 1) * 512 * 256, 2*CHN,
        b_blk + (i - 1) * CHN, tmp, CHN, 2*CHN);
    bn_partial_kernel<<<256, 256, 0, stream>>>(tmp, CHN, ps, pq);
    bn_finalize_kernel<<<1, 256, 0, stream>>>(ps, pq, CHN, mu, var);
    bn_apply_kernel<<<(NTOT*CHN)/256, 256, 0, stream>>>(
        tmp, CHN, mu, var, g_blk + (i - 1) * CHN, be_blk + (i - 1) * CHN,
        hprev, FUS_IN, fCat, FUS_IN, i * CHN, fCatB);
  }

  // fusion GEMM [16384 x 7168] @ [7168 x 1024]
  gemm_bf16_relu_kernel<<<dim3(NTOT/128, POOL_DIM/128), 256, 0, stream>>>(
      fCatB, FUS_IN, WtF, FUS_IN, b_fus, tmp, POOL_DIM, FUS_IN);
  bn_partial_kernel<<<256, 256, 0, stream>>>(tmp, POOL_DIM, ps, pq);
  bn_finalize_kernel<<<POOL_DIM/256, 256, 0, stream>>>(ps, pq, POOL_DIM, mu, var);
  bn_apply_kernel<<<(NTOT*POOL_DIM)/256, 256, 0, stream>>>(
      tmp, POOL_DIM, mu, var, g_fus, be_fus, nullptr, 0, fus, POOL_DIM, 0, nullptr);

  // top-k pooling + global max/mean/min
  wnorm_kernel<<<1, 256, 0, stream>>>(w_pool, invn);
  score_kernel<<<NTOT, 256, 0, stream>>>(fus, w_pool, invn, score);
  topk_sort_kernel<<<N_GRAPHS, 256, 0, stream>>>(score, vals, nidx);
  pool_stats_kernel<<<dim3(N_GRAPHS, POOL_DIM/256), 256, 0, stream>>>(fus, vals, nidx, gfeat);

  // prediction head (tiny: 16 rows)
  lin_relu_kernel<<<(N_GRAPHS*512)/256, 256, 0, stream>>>(gfeat, 3*POOL_DIM, W_p1, b_p1, h1a, 512, 1);
  bn16_kernel<<<2, 256, 0, stream>>>(h1a, 512, g_p1, be_p1, h1b);
  lin_relu_kernel<<<(N_GRAPHS*256)/256, 256, 0, stream>>>(h1b, 512, W_p2, b_p2, h2a, 256, 1);
  bn16_kernel<<<1, 256, 0, stream>>>(h2a, 256, g_p2, be_p2, h2b);
  lin_relu_kernel<<<1, 32, 0, stream>>>(h2b, 256, W_p3, b_p3, (float*)d_out, 2, 0);
}
